// HierarchicalSparseAttention_91147795955934
// MI455X (gfx1250) — matrix operand order repaired and checked by reading, not compiled, not run on hardware
//
#include <hip/hip_runtime.h>
#include <hip/hip_bf16.h>
#include <math.h>

// ---------------------------------------------------------------------------
// HSA forward for gfx1250 (MI455X).  All dense math via v_wmma_f32_16x16x32_f16.
// Shapes: B=1, L=1024, D=2048, H=16, HKV=4, HD=128, S=32 (C=32), TOPK=8, RD=256
// ---------------------------------------------------------------------------

typedef _Float16 v8h  __attribute__((ext_vector_type(8)));
typedef _Float16 v16h __attribute__((ext_vector_type(16)));
typedef float    v8f  __attribute__((ext_vector_type(8)));

union H16 {
  v16h v;
  v8h  h[2];
  _Float16 f[16];
  unsigned u32[8];
};

constexpr int kD   = 2048;
constexpr int kL   = 1024;
constexpr int kH   = 16;
constexpr int kHKV = 4;
constexpr int kHD  = 128;
constexpr int kS   = 32;
constexpr int kC   = 32;   // kL / kS
constexpr int kTOP = 8;
constexpr int kRD  = 256;
constexpr float kEPS = 1e-5f;

// ---------------------------------------------------------------------------
// fp32 -> fp16 conversion (weights)
// ---------------------------------------------------------------------------
__global__ void f32_to_f16(const float* __restrict__ in, _Float16* __restrict__ out, int n) {
  int i = blockIdx.x * 256 + threadIdx.x;
  if (i < n) out[i] = (_Float16)in[i];
}

// ---------------------------------------------------------------------------
// Prep: x = rms(h, g_pre); xc = rms(x, g_chunk); xr = rms(x, g_pre_retr)
// (xc and xr share the same variance of x).  One block per row (256 thr).
// ---------------------------------------------------------------------------
__global__ __launch_bounds__(256) void prep_norms(
    const float* __restrict__ hs, const float* __restrict__ g_pre,
    const float* __restrict__ g_chunk, const float* __restrict__ g_pre_retr,
    _Float16* __restrict__ x16, _Float16* __restrict__ xc16, _Float16* __restrict__ xr16)
{
  const int row = blockIdx.x;
  const int tid = threadIdx.x;
  const float* h = hs + (size_t)row * kD;
  __shared__ float red[256];

  float loc[8];
  float ss = 0.f;
#pragma unroll
  for (int i = 0; i < 8; ++i) { loc[i] = h[tid + i * 256]; ss += loc[i] * loc[i]; }
  red[tid] = ss; __syncthreads();
  for (int s = 128; s > 0; s >>= 1) { if (tid < s) red[tid] += red[tid + s]; __syncthreads(); }
  const float inv1 = rsqrtf(red[0] / (float)kD + kEPS);
  __syncthreads();

  float x[8]; float ss2 = 0.f;
#pragma unroll
  for (int i = 0; i < 8; ++i) {
    int d = tid + i * 256;
    x[i] = loc[i] * inv1 * g_pre[d];
    ss2 += x[i] * x[i];
  }
  red[tid] = ss2; __syncthreads();
  for (int s = 128; s > 0; s >>= 1) { if (tid < s) red[tid] += red[tid + s]; __syncthreads(); }
  const float inv2 = rsqrtf(red[0] / (float)kD + kEPS);

#pragma unroll
  for (int i = 0; i < 8; ++i) {
    int d = tid + i * 256;
    size_t o = (size_t)row * kD + d;
    x16[o]  = (_Float16)x[i];
    xc16[o] = (_Float16)(x[i] * inv2 * g_chunk[d]);
    xr16[o] = (_Float16)(x[i] * inv2 * g_pre_retr[d]);
  }
}

// ---------------------------------------------------------------------------
// Per-chunk mean of xc: xcm[c][d] = mean_s xc[c*32+s][d]
// ---------------------------------------------------------------------------
__global__ void chunk_mean(const _Float16* __restrict__ xc16, _Float16* __restrict__ xcm) {
  int i = blockIdx.x * 256 + threadIdx.x;       // 32*2048 elements
  int c = i >> 11, d = i & (kD - 1);
  float s = 0.f;
  for (int t = 0; t < kS; ++t) s += (float)xc16[(size_t)(c * kS + t) * kD + d];
  xcm[i] = (_Float16)(s * (1.f / (float)kS));
}

// ---------------------------------------------------------------------------
// WMMA fragment loaders (wave32 layouts per CDNA5 ISA §7.12.2)
// ---------------------------------------------------------------------------
__device__ inline v16h load_a_frag(const _Float16* __restrict__ arow, int d0, int lane) {
  // lane holds row M=lane&15; halves 0-7: K=d0+off+h, halves 8-15: K=d0+off+16+h
  const int off = (lane & 16) ? 8 : 0;
  H16 u;
  u.h[0] = *(const v8h*)(arow + d0 + off);
  u.h[1] = *(const v8h*)(arow + d0 + off + 16);
  return u.v;
}
__device__ inline v16h load_b_frag(const _Float16* __restrict__ brow, int d0, int lane) {
  // lane holds col N; halves 0-7: K=d0+off+h, halves 8-15: K=d0+off+16+h
  const int off = (lane & 16) ? 8 : 0;
  H16 u;
  u.h[0] = *(const v8h*)(brow + d0 + off);
  u.h[1] = *(const v8h*)(brow + d0 + off + 16);
  return u.v;
}

// ---------------------------------------------------------------------------
// C = A(MxK,f16) @ W(NxK,f16)^T (+ optional residual) -> fp32.
// Compile-time geometry: wave computes 16 x (NT*16); block = NWAVES waves.
// Launch geometry guarantees every tile is in range -> no divergent guards,
// EXEC stays all-1s around every WMMA (ISA requirement).
// ---------------------------------------------------------------------------
template <int N, int K, int NT, int NWAVES, bool RES>
__global__ __launch_bounds__(NWAVES * 32) void gemm_xwT(
    const _Float16* __restrict__ A, const _Float16* __restrict__ W,
    float* __restrict__ Cout, const float* __restrict__ residual)
{
  const int lane  = threadIdx.x & 31;
  const int wave  = threadIdx.x >> 5;
  const int m0    = blockIdx.x * 16;
  const int nbase = blockIdx.y * (NWAVES * NT * 16) + wave * (NT * 16);
  const int row   = m0 + (lane & 15);

  v8f acc[NT];
#pragma unroll
  for (int t = 0; t < NT; ++t)
#pragma unroll
    for (int r = 0; r < 8; ++r) acc[t][r] = 0.f;

  const _Float16* arow = A + (size_t)row * K;
  for (int d0 = 0; d0 < K; d0 += 32) {
    v16h a = load_a_frag(arow, d0, lane);
#pragma unroll
    for (int t = 0; t < NT; ++t) {
      const _Float16* wrow = W + (size_t)(nbase + t * 16 + (lane & 15)) * K;
      v16h b = load_b_frag(wrow, d0, lane);
      acc[t] = __builtin_amdgcn_wmma_f32_16x16x32_f16(false, a, false, b,
                                                      (short)0, acc[t], false, false);
    }
  }
  const int rbase = m0 + ((lane & 16) ? 8 : 0);
#pragma unroll
  for (int t = 0; t < NT; ++t) {
    int col = nbase + t * 16 + (lane & 15);
#pragma unroll
    for (int r = 0; r < 8; ++r) {
      size_t o = (size_t)(rbase + r) * N + col;
      if constexpr (RES) Cout[o] = acc[t][r] + residual[o];
      else               Cout[o] = acc[t][r];
    }
  }
}

// ---------------------------------------------------------------------------
// Row-wise RMS norm (width WID), fp32 in -> fp16 out.  block = WID threads.
// ---------------------------------------------------------------------------
template <int WID>
__global__ void rms_rows(const float* __restrict__ in, const float* __restrict__ g,
                         _Float16* __restrict__ out)
{
  const int row = blockIdx.x;
  const int tid = threadIdx.x;
  float v = in[(size_t)row * WID + tid];
  __shared__ float red[WID];
  red[tid] = v * v; __syncthreads();
  for (int s = WID / 2; s > 0; s >>= 1) { if (tid < s) red[tid] += red[tid + s]; __syncthreads(); }
  float inv = rsqrtf(red[0] / (float)WID + kEPS);
  out[(size_t)row * WID + tid] = (_Float16)(v * inv * g[tid]);
}

// ---------------------------------------------------------------------------
// V transpose: vraw (tok, kv*HD) fp32 -> vT (kv*HD, tok) fp16
// ---------------------------------------------------------------------------
__global__ void v_transpose(const float* __restrict__ vraw, _Float16* __restrict__ vT) {
  int i = blockIdx.x * 256 + threadIdx.x;       // kL * kHKV*kHD
  int tok = i >> 9, r = i & 511;
  vT[(size_t)r * kL + tok] = (_Float16)vraw[i];
}

// ---------------------------------------------------------------------------
// Causal chunk top-k + retrieval softmax.  One thread per token.
// ---------------------------------------------------------------------------
__global__ void topk_kernel(const float* __restrict__ scores,
                            int* __restrict__ idxOut, float* __restrict__ rwOut)
{
  int l = blockIdx.x * 256 + threadIdx.x;
  if (l >= kL) return;
  const float* sc = scores + (size_t)l * kC;
  const int m = l >> 5;                         // number of fully-preceding chunks
  float s[kC];
#pragma unroll
  for (int c = 0; c < kC; ++c) s[c] = (c < m) ? sc[c] * 0.0625f : -INFINITY; // 1/sqrt(256)

  unsigned used = 0u;
  int idxv[kTOP]; float selv[kTOP];
  for (int i = 0; i < kTOP; ++i) {
    float best = -INFINITY; int bi = -1;
#pragma unroll
    for (int c = 0; c < kC; ++c)
      if (!((used >> c) & 1u) && s[c] > best) { best = s[c]; bi = c; }
    if (bi < 0) { bi = __builtin_ctz(~used); best = -INFINITY; } // -inf ties: index order
    used |= 1u << bi;
    idxv[i] = bi; selv[i] = best;
  }
  // sort ascending by chunk index
  for (int i = 1; i < kTOP; ++i) {
    int ki = idxv[i]; float kvv = selv[i]; int j = i - 1;
    while (j >= 0 && idxv[j] > ki) { idxv[j + 1] = idxv[j]; selv[j + 1] = selv[j]; --j; }
    idxv[j + 1] = ki; selv[j + 1] = kvv;
  }
  float rw[kTOP];
  if (m == 0) {
    for (int i = 0; i < kTOP; ++i) rw[i] = 0.f;
  } else {
    float mx = -INFINITY;
    for (int i = 0; i < kTOP; ++i) mx = fmaxf(mx, selv[i]);
    float sum = 0.f;
    for (int i = 0; i < kTOP; ++i) {
      float e = (selv[i] == -INFINITY) ? 0.f : __expf(selv[i] - mx);
      rw[i] = e; sum += e;
    }
    for (int i = 0; i < kTOP; ++i) rw[i] /= sum;
  }
  for (int i = 0; i < kTOP; ++i) { idxOut[l * kTOP + i] = idxv[i]; rwOut[l * kTOP + i] = rw[i]; }
}

// ---------------------------------------------------------------------------
// Gathered attention: block = token, wave = kv-head.
// QK^T (WMMA, M=4 padded to 16) -> per-chunk softmax * rw -> PV (WMMA).
// ---------------------------------------------------------------------------
__global__ __launch_bounds__(128) void attn_kernel(
    const _Float16* __restrict__ q16,   // (L, H, HD)
    const _Float16* __restrict__ k16,   // (L, HKV, HD)
    const _Float16* __restrict__ vT16,  // (HKV*HD, L)
    const int* __restrict__ idx, const float* __restrict__ rw,
    _Float16* __restrict__ ao16)        // (L, H, HD)
{
  const int tok  = blockIdx.x;
  const int kv   = threadIdx.x >> 5;
  const int lane = threadIdx.x & 31;
  const float scale = 0.0883883476f;    // 1/sqrt(128)

  __shared__ float sP[kHKV][kTOP * kS]; // per-wave scores/probs (4 x 256)
  __shared__ int   sIdx[kHKV][kTOP];
  __shared__ float sRw[kHKV][kTOP];
  if (lane < kTOP) {
    sIdx[kv][lane] = idx[tok * kTOP + lane];
    sRw[kv][lane]  = rw[tok * kTOP + lane];
  }
  __syncthreads();

  // --- Q fragments: 4 k-steps over HD=128; rows 4..15 zero-padded ---
  v16h aq[4];
  {
    const int g = lane & 15;
    const _Float16* qp = q16 + (size_t)tok * (kH * kHD) + (kv * 4 + g) * kHD;
    const int off = (lane & 16) ? 8 : 0;
#pragma unroll
    for (int ks = 0; ks < 4; ++ks) {
      H16 u;
      if (g < 4) {
        u.h[0] = *(const v8h*)(qp + ks * 32 + off);
        u.h[1] = *(const v8h*)(qp + ks * 32 + off + 16);
      } else {
#pragma unroll
        for (int j = 0; j < 8; ++j) u.u32[j] = 0u;
      }
      aq[ks] = u.v;
    }
  }

  // --- QK^T: 16 n-tiles of 16 keys; write rows 0..3 (4x256 scores) ---
  for (int t = 0; t < 16; ++t) {
    const int ci   = t >> 1;
    const int base = sIdx[kv][ci] * kS + (t & 1) * 16;   // global token of key block
    v8f acc;
#pragma unroll
    for (int r = 0; r < 8; ++r) acc[r] = 0.f;
#pragma unroll
    for (int ks = 0; ks < 4; ++ks) {
      const _Float16* kp = k16 + (size_t)(base + (lane & 15)) * (kHKV * kHD) + kv * kHD;
      v16h b = load_b_frag(kp, ks * 32, lane);
      acc = __builtin_amdgcn_wmma_f32_16x16x32_f16(false, aq[ks], false, b,
                                                   (short)0, acc, false, false);
    }
    if (lane < 16) {
#pragma unroll
      for (int g = 0; g < 4; ++g) sP[kv][g * 256 + t * 16 + lane] = acc[g];
    }
  }
  __syncthreads();

  // --- per-chunk softmax over S=32, times retrieval weight ---
  for (int g = 0; g < 4; ++g)
    for (int ci = 0; ci < kTOP; ++ci) {
      float val = sP[kv][g * 256 + ci * kS + lane] * scale;
      float mx = val;
#pragma unroll
      for (int o = 16; o > 0; o >>= 1) mx = fmaxf(mx, __shfl_xor(mx, o, 32));
      float e = __expf(val - mx);
      float sum = e;
#pragma unroll
      for (int o = 16; o > 0; o >>= 1) sum += __shfl_xor(sum, o, 32);
      sP[kv][g * 256 + ci * kS + lane] = e * (sRw[kv][ci] / sum);
    }
  __syncthreads();

  // --- P fragments: one per selected chunk (K-step = one chunk of 32 keys) ---
  v16h ap[kTOP];
  {
    const int g = lane & 15;
#pragma unroll
    for (int ci = 0; ci < kTOP; ++ci) {
      H16 u;
      if (g < 4) {
#pragma unroll
        for (int h = 0; h < 16; ++h) {
          int kk = ((h < 8) ? h : h + 8) + ((lane & 16) ? 8 : 0);
          u.f[h] = (_Float16)sP[kv][g * 256 + ci * kS + kk];
        }
      } else {
#pragma unroll
        for (int j = 0; j < 8; ++j) u.u32[j] = 0u;
      }
      ap[ci] = u.v;
    }
  }

  // --- PV: 8 n-tiles over HD, K = 8 chunks ---
  for (int nt = 0; nt < 8; ++nt) {
    v8f acc;
#pragma unroll
    for (int r = 0; r < 8; ++r) acc[r] = 0.f;
#pragma unroll
    for (int ci = 0; ci < kTOP; ++ci) {
      const int base = sIdx[kv][ci] * kS;
      const _Float16* vp = vT16 + (size_t)(kv * kHD + nt * 16 + (lane & 15)) * kL + base;
      H16 u;
      const int off = (lane & 16) ? 8 : 0;
      u.h[0] = *(const v8h*)(vp + off);
      u.h[1] = *(const v8h*)(vp + off + 16);
      acc = __builtin_amdgcn_wmma_f32_16x16x32_f16(false, ap[ci], false, u.v,
                                                   (short)0, acc, false, false);
    }
    if (lane < 16) {
#pragma unroll
      for (int g = 0; g < 4; ++g)
        ao16[(size_t)tok * (kH * kHD) + (kv * 4 + g) * kHD + nt * 16 + lane] = (_Float16)acc[g];
    }
  }
}

// ---------------------------------------------------------------------------
// Host launcher
// ---------------------------------------------------------------------------
extern "C" void kernel_launch(void* const* d_in, const int* in_sizes, int n_in,
                              void* d_out, int out_size, void* d_ws, size_t ws_size,
                              hipStream_t stream)
{
  const float* hs      = (const float*)d_in[0];
  const float* w_k     = (const float*)d_in[1];
  const float* w_v     = (const float*)d_in[2];
  const float* w_lmk   = (const float*)d_in[3];
  const float* w_qr    = (const float*)d_in[4];
  const float* w_q     = (const float*)d_in[5];
  const float* w_o     = (const float*)d_in[6];
  const float* g_chunk = (const float*)d_in[7];
  const float* g_knorm = (const float*)d_in[8];
  const float* g_preR  = (const float*)d_in[9];
  const float* g_qretr = (const float*)d_in[10];
  const float* g_lmk   = (const float*)d_in[11];
  const float* g_pre   = (const float*)d_in[12];
  const float* g_qnorm = (const float*)d_in[13];

  char* ws = (char*)d_ws;
  size_t off = 0;
  auto alloc = [&](size_t bytes) -> void* {
    off = (off + 255) & ~(size_t)255;
    void* p = ws + off;
    off += bytes;
    return p;
  };

  _Float16* x16   = (_Float16*)alloc((size_t)kL * kD * 2);
  _Float16* xc16  = (_Float16*)alloc((size_t)kL * kD * 2);
  _Float16* xr16  = (_Float16*)alloc((size_t)kL * kD * 2);
  _Float16* xcm16 = (_Float16*)alloc((size_t)kC * kD * 2);
  _Float16* wk16  = (_Float16*)alloc((size_t)(kHKV * kHD) * kD * 2);
  _Float16* wv16  = (_Float16*)alloc((size_t)(kHKV * kHD) * kD * 2);
  _Float16* wlm16 = (_Float16*)alloc((size_t)kRD * kD * 2);
  _Float16* wqr16 = (_Float16*)alloc((size_t)kRD * kD * 2);
  _Float16* wq16  = (_Float16*)alloc((size_t)(kH * kHD) * kD * 2);
  _Float16* wo16  = (_Float16*)alloc((size_t)kD * (kH * kHD) * 2);
  float*    kraw  = (float*)   alloc((size_t)kL * kHKV * kHD * 4);
  float*    vraw  = (float*)   alloc((size_t)kL * kHKV * kHD * 4);
  _Float16* k16   = (_Float16*)alloc((size_t)kL * kHKV * kHD * 2);
  _Float16* vT16  = (_Float16*)alloc((size_t)kHKV * kHD * kL * 2);
  float*    qrraw = (float*)   alloc((size_t)kL * kRD * 4);
  _Float16* qr16  = (_Float16*)alloc((size_t)kL * kRD * 2);
  float*    lmraw = (float*)   alloc((size_t)kC * kRD * 4);
  _Float16* lm16  = (_Float16*)alloc((size_t)kC * kRD * 2);
  float*    scores= (float*)   alloc((size_t)kL * kC * 4);
  int*      idxb  = (int*)     alloc((size_t)kL * kTOP * 4);
  float*    rwb   = (float*)   alloc((size_t)kL * kTOP * 4);
  float*    qraw  = (float*)   alloc((size_t)kL * kD * 4);
  _Float16* q16   = (_Float16*)alloc((size_t)kL * kD * 2);
  _Float16* ao16  = (_Float16*)alloc((size_t)kL * kD * 2);
  (void)ws_size; (void)in_sizes; (void)n_in; (void)out_size;

  // 1) weight conversions
  auto conv = [&](const float* src, _Float16* dst, int n) {
    f32_to_f16<<<(n + 255) / 256, 256, 0, stream>>>(src, dst, n);
  };
  conv(w_k,   wk16,  kHKV * kHD * kD);
  conv(w_v,   wv16,  kHKV * kHD * kD);
  conv(w_lmk, wlm16, kRD * kD);
  conv(w_qr,  wqr16, kRD * kD);
  conv(w_q,   wq16,  kH * kHD * kD);
  conv(w_o,   wo16,  kD * kH * kHD);

  // 2) fused triple-RMS prep
  prep_norms<<<kL, 256, 0, stream>>>(hs, g_pre, g_chunk, g_preR, x16, xc16, xr16);

  // 3) chunk means for landmarks
  chunk_mean<<<(kC * kD) / 256, 256, 0, stream>>>(xc16, xcm16);

  // 4) K / V projections (WMMA)
  gemm_xwT<512, 2048, 4, 4, false><<<dim3(kL / 16, 2), 128, 0, stream>>>(xc16, wk16, kraw, nullptr);
  gemm_xwT<512, 2048, 4, 4, false><<<dim3(kL / 16, 2), 128, 0, stream>>>(xc16, wv16, vraw, nullptr);
  rms_rows<128><<<kL * kHKV, 128, 0, stream>>>(kraw, g_knorm, k16);
  v_transpose<<<(kL * kHKV * kHD) / 256, 256, 0, stream>>>(vraw, vT16);

  // 5) retrieval queries and landmarks
  gemm_xwT<256, 2048, 4, 4, false><<<dim3(kL / 16, 1), 128, 0, stream>>>(xr16, wqr16, qrraw, nullptr);
  rms_rows<256><<<kL, 256, 0, stream>>>(qrraw, g_qretr, qr16);
  gemm_xwT<256, 2048, 4, 4, false><<<dim3(kC / 16, 1), 128, 0, stream>>>(xcm16, wlm16, lmraw, nullptr);
  rms_rows<256><<<kC, 256, 0, stream>>>(lmraw, g_lmk, lm16);

  // 6) retrieval scores (L x C) and top-k
  gemm_xwT<32, 256, 1, 2, false><<<dim3(kL / 16, 1), 64, 0, stream>>>(qr16, lm16, scores, nullptr);
  topk_kernel<<<kL / 256, 256, 0, stream>>>(scores, idxb, rwb);

  // 7) Q projection + per-head RMS
  gemm_xwT<2048, 2048, 8, 4, false><<<dim3(kL / 16, 4), 128, 0, stream>>>(x16, wq16, qraw, nullptr);
  rms_rows<128><<<kL * kH, 128, 0, stream>>>(qraw, g_qnorm, q16);

  // 8) gathered sparse attention (WMMA)
  attn_kernel<<<kL, 128, 0, stream>>>(q16, k16, vT16, idxb, rwb, ao16);

  // 9) output projection + residual into d_out (fp32)
  gemm_xwT<2048, 2048, 8, 4, true><<<dim3(kL / 16, 4), 128, 0, stream>>>(ao16, wo16, (float*)d_out, hs);
}
